// DualCrossAttention_45071386804794
// MI455X (gfx1250) — matrix operand order repaired and checked by reading, not compiled, not run on hardware
//
#include <hip/hip_runtime.h>
#include <hip/hip_bf16.h>

#define SEQ  2048
#define DIM  512
#define NH   8
#define HDIM 64
#define NB   4
static constexpr float SCALE = 0.125f; // 1/sqrt(64)

typedef __attribute__((ext_vector_type(16))) __bf16 bf16x16;
typedef __attribute__((ext_vector_type(8)))  float  floatx8;
typedef __attribute__((ext_vector_type(4)))  unsigned int v4u;
typedef __attribute__((ext_vector_type(8)))  int          v8i;
typedef __attribute__((ext_vector_type(4)))  int          v4i;

union Frag { bf16x16 bf; uint4 u[2]; };
union U8   { uint4 u; unsigned short s[8]; };

__device__ __forceinline__ unsigned short f2bf(float f) {
  union { float f; unsigned u; } v; v.f = f;
  unsigned r = v.u + 0x7FFFu + ((v.u >> 16) & 1u);   // round-to-nearest-even
  return (unsigned short)(r >> 16);
}

__device__ __forceinline__ floatx8 wmma_bf16(const Frag& a, const Frag& b, floatx8 c) {
  // D = A(16x32 bf16) x B(32x16 bf16) + C(16x16 f32)
  return __builtin_amdgcn_wmma_f32_16x16x32_bf16(false, a.bf, false, b.bf,
                                                 (short)0, c, false, false);
}

// ---------------------------------------------------------------------------
// f32 -> bf16 conversion (vectorized)
// ---------------------------------------------------------------------------
__global__ void cvt_bf16_kernel(const float* __restrict__ X,
                                unsigned short* __restrict__ Y, int n) {
  int i = (blockIdx.x * blockDim.x + threadIdx.x) * 4;
  if (i >= n) return;
  float4 f = *(const float4*)(X + i);
  ushort4 o;
  o.x = f2bf(f.x); o.y = f2bf(f.y); o.z = f2bf(f.z); o.w = f2bf(f.w);
  *(ushort4*)(Y + i) = o;
}

// context [B*SC, 2*CROSS] -> image bf16 [B*SC,512], mask bf16 [B*SC,512]
__global__ void split_ctx_kernel(const float* __restrict__ C,
                                 unsigned short* __restrict__ img,
                                 unsigned short* __restrict__ msk) {
  int i = (blockIdx.x * blockDim.x + threadIdx.x) * 4;  // over B*SC*512
  const int s = i >> 9;
  const int c = i & 511;
  float4 fi = *(const float4*)(C + (size_t)s * 1024 + c);
  float4 fm = *(const float4*)(C + (size_t)s * 1024 + 512 + c);
  ushort4 oi, om;
  oi.x = f2bf(fi.x); oi.y = f2bf(fi.y); oi.z = f2bf(fi.z); oi.w = f2bf(fi.w);
  om.x = f2bf(fm.x); om.y = f2bf(fm.y); om.z = f2bf(fm.z); om.w = f2bf(fm.w);
  *(ushort4*)(img + i) = oi;
  *(ushort4*)(msk + i) = om;
}

// ---------------------------------------------------------------------------
// Pack W [K=512, N=512] f32 row-major into WMMA B-fragment tiles:
// tile (nb,kb) = 32 lanes x 16 bf16 contiguous; lane n<16: K=kb*32+{0..7,16..23} of
// column nb*16+n; lane n>=16: K=kb*32+{8..15,24..31}.  One wave packs one tile.
// ---------------------------------------------------------------------------
__global__ void pack_w_kernel(const float* __restrict__ W,
                              unsigned short* __restrict__ Wp) {
  const int w    = threadIdx.x >> 5;
  const int lane = threadIdx.x & 31;
  const int l16  = lane & 15, hv = lane >> 4;
  const int tile = blockIdx.x * 8 + w;       // 512 tiles total (grid = 64)
  const int nb = tile >> 4, kb = tile & 15;
  const int n  = nb * 16 + l16;
  const int k0 = kb * 32 + hv * 8;
  U8 lo, hi8;
#pragma unroll
  for (int j = 0; j < 8; ++j) lo.s[j]  = f2bf(W[(size_t)(k0 + j)     * DIM + n]);
#pragma unroll
  for (int j = 0; j < 8; ++j) hi8.s[j] = f2bf(W[(size_t)(k0 + 16 + j) * DIM + n]);
  uint4* dst = (uint4*)(Wp + (size_t)tile * 512 + lane * 16);
  dst[0] = lo.u;
  dst[1] = hi8.u;
}

// ---------------------------------------------------------------------------
// GEMM: C[M,512] = A[M,512](bf16) @ W(packed bf16).  Wave = 32 rows x 64 cols
// (2 A-frags x 4 B-frags = 8 WMMAs per 32-deep k-step; A reused 4x, B 2x),
// register double-buffered so the next k-step's 12 loads overlap 8 WMMAs.
// ---------------------------------------------------------------------------
__global__ __launch_bounds__(256) void gemm_kernel(
    const unsigned short* __restrict__ A,
    const unsigned short* __restrict__ Wp,
    unsigned short* __restrict__ Obf,
    float* __restrict__ Of,
    const float* __restrict__ bias, int M) {
  const int w    = threadIdx.x >> 5;
  const int lane = threadIdx.x & 31;
  const int l16  = lane & 15, hv = lane >> 4;
  const int r0   = blockIdx.x * 256 + w * 32;
  const int n0   = blockIdx.y * 64;

  floatx8 acc[2][4] = {};
  const unsigned short* ap    = A + (size_t)(r0 + l16) * DIM;
  const unsigned short* wbase = Wp + ((size_t)(n0 >> 4) * 16 * 32 + lane) * 16;

  Frag a[2][2];     // [buf][row-half]
  Frag b[2][4];     // [buf][n-tile]
  auto loadk = [&](int kb, int buf) {
#pragma unroll
    for (int rh = 0; rh < 2; ++rh) {
      const unsigned short* arp = ap + (size_t)rh * 16 * DIM + kb * 32 + hv * 8;
      a[buf][rh].u[0] = *(const uint4*)(arp);
      a[buf][rh].u[1] = *(const uint4*)(arp + 16);
    }
#pragma unroll
    for (int t = 0; t < 4; ++t) {
      const unsigned short* bp = wbase + t * 8192 + kb * 512;
      b[buf][t].u[0] = *(const uint4*)(bp);
      b[buf][t].u[1] = *(const uint4*)(bp + 8);
    }
  };

  loadk(0, 0);
#pragma unroll
  for (int kb = 0; kb < 16; ++kb) {
    const int cur = kb & 1;
    if (kb < 15) loadk(kb + 1, cur ^ 1);   // prefetch next k-step into regs
#pragma unroll
    for (int rh = 0; rh < 2; ++rh)
#pragma unroll
      for (int t = 0; t < 4; ++t)
        acc[rh][t] = wmma_bf16(a[cur][rh], b[cur][t], acc[rh][t]);
  }

#pragma unroll
  for (int rh = 0; rh < 2; ++rh) {
#pragma unroll
    for (int t = 0; t < 4; ++t) {
#pragma unroll
      for (int v = 0; v < 8; ++v) {
        const int row = r0 + rh * 16 + v + hv * 8; // C-tile: lanes<16 M=v, else M=v+8
        const int col = n0 + t * 16 + l16;
        float val = acc[rh][t][v];
        if (bias) val += bias[col];
        if (Of) Of[(size_t)row * DIM + col] = val;
        else    Obf[(size_t)row * DIM + col] = f2bf(val);
      }
    }
  }
}

// ---------------------------------------------------------------------------
// Flash attention: Q,K,V are bf16 [B, S, 512] with head h at columns h*64..
// WG = 128 queries (8 waves x 16 rows), online softmax over key chunks of 32.
// K chunk is DMA'd into LDS by the Tensor Data Mover (tensor_load_to_lds,
// TENSORcnt); V chunk is staged transposed by the workgroup.
// ---------------------------------------------------------------------------
__global__ __launch_bounds__(256, 2) void attn_kernel(
    const unsigned short* __restrict__ Q,
    const unsigned short* __restrict__ K,
    const unsigned short* __restrict__ V,
    unsigned short* __restrict__ O) {
  __shared__ __align__(16) unsigned short KT[32 * HDIM];   // 4 KB  K chunk [k][d] (TDM)
  __shared__ __align__(16) unsigned short VT[HDIM * 32];   // 4 KB  V^T chunk [d][k]
  __shared__ __align__(16) unsigned short PB[8 * 16 * 32]; // 8 KB  wave-private P tiles

  const int b    = blockIdx.x >> 3;
  const int h    = blockIdx.x & 7;
  const int w    = threadIdx.x >> 5;
  const int lane = threadIdx.x & 31;
  const int l16  = lane & 15, hv = lane >> 4;
  const int q0   = blockIdx.y * 128 + w * 16;

  const size_t headoff = ((size_t)b * SEQ) * DIM + h * HDIM;
  const unsigned ldsKT = (unsigned)(uintptr_t)&KT[0];      // LDS byte offset

  // Q A-fragments for the two K-steps of the d=64 dot product
  Frag qa[2];
  {
    const unsigned short* qp = Q + headoff + (size_t)(q0 + l16) * DIM;
#pragma unroll
    for (int s = 0; s < 2; ++s) {
      qa[s].u[0] = *(const uint4*)(qp + s * 32 + hv * 8);
      qa[s].u[1] = *(const uint4*)(qp + s * 32 + hv * 8 + 16);
    }
  }

  float m[8], l[8];
  floatx8 o[4];
  const floatx8 fz = {};
#pragma unroll
  for (int v = 0; v < 8; ++v) { m[v] = -3.0e38f; l[v] = 0.f; }
#pragma unroll
  for (int n = 0; n < 4; ++n) o[n] = fz;

  for (int kc = 0; kc < SEQ; kc += 32) {
    __syncthreads();   // prior chunk's LDS reads complete

    // --- TDM: DMA K tile [32 keys x 64 d] bf16 -> LDS (wave 0 issues) ---
    if (w == 0) {
      const unsigned long long gaddr =
          (unsigned long long)(uintptr_t)(K + headoff + (size_t)kc * DIM);
      v4u g0 = { 1u,                                   // count=1 (valid user D#)
                 ldsKT,                                // lds_addr
                 (unsigned)(gaddr & 0xFFFFFFFFu),      // global_addr[31:0]
                 (unsigned)((gaddr >> 32) & 0x01FFFFFFu) | (2u << 30) }; // type=2
      v8i g1 = { (int)(1u << 16),                      // data_size=1 (2 bytes)
                 (int)(64u << 16),                     // tensor_dim0 = 64
                 (int)(32u << 16),                     // tensor_dim1 = 32
                 (int)(64u << 16),                     // tile_dim0 = 64
                 (int)(32u),                           // tile_dim1 = 32, tile_dim2 = 0
                 (int)(DIM),                           // tensor_dim0_stride = 512
                 0, 0 };
      v4i gz4 = { 0, 0, 0, 0 };
      v8i gz8 = { 0, 0, 0, 0, 0, 0, 0, 0 };
      __builtin_amdgcn_tensor_load_to_lds(g0, g1, gz4, gz4, gz8, 0);
      __builtin_amdgcn_s_wait_tensorcnt(0);
    }

    // --- stage V^T [d][k] cooperatively (256 thr x 8 elems) ---
    {
      const int kk = threadIdx.x >> 3;
      const int dg = (threadIdx.x & 7) * 8;
      U8 tmp;
      tmp.u = *(const uint4*)(V + headoff + (size_t)(kc + kk) * DIM + dg);
#pragma unroll
      for (int j = 0; j < 8; ++j) VT[(dg + j) * 32 + kk] = tmp.s[j];
    }
    __syncthreads();

    if (kc + 32 < SEQ)
      __builtin_prefetch(V + headoff + (size_t)(kc + 32 + (threadIdx.x >> 3)) * DIM, 0, 1);

    // --- scores: gather all 4 K^T B-fragments, then 4 WMMAs ---
    Frag kf[4];
#pragma unroll
    for (int t = 0; t < 2; ++t)
#pragma unroll
      for (int ds = 0; ds < 2; ++ds) {
        const unsigned short* kp = KT + (t * 16 + l16) * HDIM + ds * 32 + hv * 8;
        kf[t * 2 + ds].u[0] = *(const uint4*)(kp);
        kf[t * 2 + ds].u[1] = *(const uint4*)(kp + 16);
      }
    floatx8 sc[2];
#pragma unroll
    for (int t = 0; t < 2; ++t) {
      floatx8 s = fz;
      s = wmma_bf16(qa[0], kf[t * 2 + 0], s);
      s = wmma_bf16(qa[1], kf[t * 2 + 1], s);
#pragma unroll
      for (int v = 0; v < 8; ++v) s[v] *= SCALE;
      sc[t] = s;
    }

    // --- online softmax: row = v + 8*hv; reductions stay in 16-lane halves ---
    float p0[8], p1[8];
#pragma unroll
    for (int v = 0; v < 8; ++v) {
      float rm = fmaxf(sc[0][v], sc[1][v]);
      rm = fmaxf(rm, __shfl_xor(rm, 1));
      rm = fmaxf(rm, __shfl_xor(rm, 2));
      rm = fmaxf(rm, __shfl_xor(rm, 4));
      rm = fmaxf(rm, __shfl_xor(rm, 8));
      float mn   = fmaxf(m[v], rm);
      float corr = __expf(m[v] - mn);
      m[v] = mn;
      p0[v] = __expf(sc[0][v] - mn);
      p1[v] = __expf(sc[1][v] - mn);
      float rs = p0[v] + p1[v];
      rs += __shfl_xor(rs, 1);
      rs += __shfl_xor(rs, 2);
      rs += __shfl_xor(rs, 4);
      rs += __shfl_xor(rs, 8);
      l[v] = l[v] * corr + rs;
#pragma unroll
      for (int n = 0; n < 4; ++n) o[n][v] *= corr;
    }

    // --- bounce P (C-tile layout) through LDS into A-fragment layout ---
    unsigned short* pb = PB + w * 512;
#pragma unroll
    for (int v = 0; v < 8; ++v) {
      const int row = v + hv * 8;
      pb[row * 32 + l16]      = f2bf(p0[v]);
      pb[row * 32 + 16 + l16] = f2bf(p1[v]);
    }
    Frag pa;
    pa.u[0] = *(const uint4*)(pb + l16 * 32 + hv * 8);
    pa.u[1] = *(const uint4*)(pb + l16 * 32 + hv * 8 + 16);

    // --- O += P @ V : gather all 4 V^T B-fragments, then 4 WMMAs ---
    Frag vf[4];
#pragma unroll
    for (int n = 0; n < 4; ++n) {
      const unsigned short* vp = VT + (n * 16 + l16) * 32 + hv * 8;
      vf[n].u[0] = *(const uint4*)(vp);
      vf[n].u[1] = *(const uint4*)(vp + 16);
    }
#pragma unroll
    for (int n = 0; n < 4; ++n) o[n] = wmma_bf16(pa, vf[n], o[n]);
  }

  // --- normalize + merge heads, bf16 out ---
#pragma unroll
  for (int n = 0; n < 4; ++n) {
#pragma unroll
    for (int v = 0; v < 8; ++v) {
      const int row = v + hv * 8;
      const float val = o[n][v] / l[v];
      O[headoff + (size_t)(q0 + row) * DIM + n * 16 + l16] = f2bf(val);
    }
  }
}

// ---------------------------------------------------------------------------
extern "C" void kernel_launch(void* const* d_in, const int* in_sizes, int n_in,
                              void* d_out, int out_size, void* d_ws, size_t ws_size,
                              hipStream_t stream) {
  const float* x   = (const float*)d_in[0];
  const float* ctx = (const float*)d_in[1];
  const float* Wq  = (const float*)d_in[2];
  const float* Wk  = (const float*)d_in[3];
  const float* Wv  = (const float*)d_in[4];
  const float* Wqm = (const float*)d_in[5];
  const float* Wki = (const float*)d_in[6];
  const float* Wvi = (const float*)d_in[7];
  const float* Wo  = (const float*)d_in[8];
  const float* bo  = (const float*)d_in[9];
  float* out = (float*)d_out;

  char* ws = (char*)d_ws;
  const size_t T = (size_t)NB * SEQ * DIM * sizeof(unsigned short); // 8 MB each
  unsigned short* xb  = (unsigned short*)(ws + 0 * T);
  unsigned short* img = (unsigned short*)(ws + 1 * T);
  unsigned short* msk = (unsigned short*)(ws + 2 * T);
  unsigned short* qb  = (unsigned short*)(ws + 3 * T);
  unsigned short* qmb = (unsigned short*)(ws + 4 * T);
  unsigned short* kib = (unsigned short*)(ws + 5 * T);
  unsigned short* vib = (unsigned short*)(ws + 6 * T);
  unsigned short* dub = (unsigned short*)(ws + 7 * T);
  unsigned short* kdb = (unsigned short*)(ws + 8 * T);
  unsigned short* vdb = (unsigned short*)(ws + 9 * T);
  unsigned short* a2b = (unsigned short*)(ws + 10 * T);
  const size_t WT = (size_t)DIM * DIM * sizeof(unsigned short);     // 512 KB each
  unsigned short* wp[7];
  for (int i = 0; i < 7; ++i) wp[i] = (unsigned short*)(ws + 11 * T + i * WT);

  const int M  = NB * SEQ;     // 8192
  const int NE = M * DIM;      // 4194304 elements

  // 0) precision conversion + weight packing
  cvt_bf16_kernel<<<NE / 4 / 256, 256, 0, stream>>>(x, xb, NE);
  split_ctx_kernel<<<NE / 4 / 256, 256, 0, stream>>>(ctx, img, msk);
  const float* wsrc[7] = {Wq, Wk, Wv, Wqm, Wki, Wvi, Wo};
  for (int i = 0; i < 7; ++i)
    pack_w_kernel<<<64, 256, 0, stream>>>(wsrc[i], wp[i]);

  dim3 gg(M / 256, 8), gb(256);
  // 1) projections
  gemm_kernel<<<gg, gb, 0, stream>>>(xb,  wp[0], qb,  nullptr, nullptr, M); // q
  gemm_kernel<<<gg, gb, 0, stream>>>(msk, wp[3], qmb, nullptr, nullptr, M); // qm
  gemm_kernel<<<gg, gb, 0, stream>>>(img, wp[4], kib, nullptr, nullptr, M); // ki
  gemm_kernel<<<gg, gb, 0, stream>>>(img, wp[5], vib, nullptr, nullptr, M); // vi

  dim3 ag(NB * NH, SEQ / 128);
  // 2) attention 1 -> dual (merged heads)
  attn_kernel<<<ag, gb, 0, stream>>>(qmb, kib, vib, dub);
  // 3) kd, vd projections of dual
  gemm_kernel<<<gg, gb, 0, stream>>>(dub, wp[1], kdb, nullptr, nullptr, M);
  gemm_kernel<<<gg, gb, 0, stream>>>(dub, wp[2], vdb, nullptr, nullptr, M);
  // 4) attention 2
  attn_kernel<<<ag, gb, 0, stream>>>(qb, kdb, vdb, a2b);
  // 5) out = a2 @ Wo + bo  (f32)
  gemm_kernel<<<gg, gb, 0, stream>>>(a2b, wp[6], nullptr, out, bo, M);
}
